// Encoder_38912403702377
// MI455X (gfx1250) — matrix operand order repaired and checked by reading, not compiled, not run on hardware
//
#include <hip/hip_runtime.h>

// ---------------------------------------------------------------------------
// DA-RNN input-attention encoder for MI455X (gfx1250, wave32, WMMA).
// B=512, T-1=99, N=128, H=256.
//
//   1) prep_weights : f32 -> f16 repack of W1 (padded to 112x512) and
//                     Wcat = [W_ih | W_hh] (1024x384), stored out-major so
//                     WMMA B-fragments load as contiguous half2 pairs.
//   2) z2_kernel    : time-invariant z2[b,n,s] = sum_t in[b,t,n]*W2[s,t]+b2[s]
//                     (LDS-tiled fp32, rows padded to 112 with zeros)
//   3) encoder_recurrent : persistent scan. 32 workgroups x 512 thr (16 waves),
//                     16 batch rows per workgroup, h/c state in LDS/registers,
//                     both per-step GEMMs via v_wmma_f32_16x16x32_f16.
// ---------------------------------------------------------------------------

#define BATCH   512
#define TSTEPS  99
#define TPAD    112          // 7 * 16 WMMA tiles (z2 row stride, z1 tile cols)
#define NFEAT   128
#define HID     256
#define H2      512          // 2*H  (K of z1 GEMM)
#define KCAT    384          // N + H (K of gate GEMM)
#define GOUT    1024         // 4*H
#define BT      16           // batch tile (WMMA M)
#define NTHR    512          // threads per workgroup (16 wave32 waves)

// LDS row strides chosen so (stride in words) % 16 == 4 -> rows land in
// distinct 4-word bank groups (64-bank LDS), and 16B alignment is preserved.
#define GSLD    1028         // gates f32 stride
#define HCLD    520          // [h|c] f16 stride
#define WHLD    392          // [w|h] f16 stride
#define TPADL   116          // z1 f32 stride

typedef __attribute__((ext_vector_type(16))) _Float16 frag16;
typedef __attribute__((ext_vector_type(8)))  float    fragf32;

__device__ __forceinline__ fragf32 wmma_f16(frag16 a, frag16 b, fragf32 c) {
  // (neg_a, A, neg_b, B, c_mod, C, reuse_a, reuse_b)
  return __builtin_amdgcn_wmma_f32_16x16x32_f16(false, a, false, b,
                                                (short)0, c, false, false);
}

// A fragment: 16x32 f16, source row-major [M x K] (ISA 7.12.2 layout).
// lane L: M = L%16; half = L/16; VGPR v<4 -> K = half*8 + 2v,
// v>=4 -> K = 16 + half*8 + 2(v-4).  Pairs are contiguous in K.
__device__ __forceinline__ frag16 load_a_rowmajor(const _Float16* __restrict__ base,
                                                  int ld, int kbase) {
  const int lane = threadIdx.x & 31;
  const int m = lane & 15, hs = lane >> 4;
  const _Float16* row = base + m * ld + kbase + hs * 8;
  frag16 a;
#pragma unroll
  for (int v = 0; v < 8; ++v) {
    const int k = ((v & 4) << 2) + 2 * (v & 3);  // v<4: 2v ; v>=4: 16+2(v-4)
    a[2 * v]     = row[k];
    a[2 * v + 1] = row[k + 1];
  }
  return a;
}

// B fragment: 32x16 f16 (K x N). Source stored out-major: W[n][k], stride ld.
// lane L: N = L%16; half = L/16; VGPR v<4 -> K = half*8 + 2v,
// v>=4 -> K = 16 + half*8 + 2(v-4).  Pairs are contiguous in K.
__device__ __forceinline__ frag16 load_b_outmajor(const _Float16* __restrict__ base,
                                                  int ld, int n0, int kbase) {
  const int lane = threadIdx.x & 31;
  const int n = lane & 15, hs = lane >> 4;
  const _Float16* row = base + (size_t)(n0 + n) * ld + kbase + hs * 8;
  frag16 b;
#pragma unroll
  for (int v = 0; v < 8; ++v) {
    const int k = ((v & 4) << 2) + 2 * (v & 3);
    b[2 * v]     = row[k];
    b[2 * v + 1] = row[k + 1];
  }
  return b;
}

__device__ __forceinline__ float sigmoidf(float x) {
  return 1.0f / (1.0f + __expf(-x));
}

// ---------------------------------------------------------------------------
// Kernel 1: repack weights to f16 (out-major B matrices).
// ---------------------------------------------------------------------------
__global__ __launch_bounds__(256) void prep_weights(
    const float* __restrict__ W1, const float* __restrict__ W_ih,
    const float* __restrict__ W_hh, _Float16* __restrict__ W1h,
    _Float16* __restrict__ Wc16) {
  const int i = blockIdx.x * 256 + threadIdx.x;
  const int N1 = TPAD * H2;
  const int N2 = GOUT * KCAT;
  if (i < N1) {
    const int r = i >> 9, k = i & (H2 - 1);
    W1h[i] = (r < TSTEPS) ? (_Float16)W1[r * H2 + k] : (_Float16)0.0f;
  } else if (i < N1 + N2) {
    const int j = i - N1;
    const int r = j / KCAT, k = j - r * KCAT;
    const float v = (k < NFEAT) ? W_ih[r * NFEAT + k]
                                : W_hh[r * HID + (k - NFEAT)];
    Wc16[j] = (_Float16)v;
  }
}

// ---------------------------------------------------------------------------
// Kernel 2: z2[b,n,s] = sum_t input[b,t,n] * W2[s,t] + b2[s]
// stored with row stride TPAD; columns 99..111 zero-filled so the recurrent
// kernel can run a float4 tanh-dot over the full padded row (w3 pad == 0).
// ---------------------------------------------------------------------------
__global__ __launch_bounds__(256) void z2_kernel(
    const float* __restrict__ input, const float* __restrict__ W2,
    const float* __restrict__ b2, float* __restrict__ z2) {
  extern __shared__ char smem_z2[];
  float* in_s = (float*)smem_z2;              // TSTEPS * NFEAT
  float* w2_s = in_s + TSTEPS * NFEAT;        // TSTEPS * TSTEPS
  float* b2_s = w2_s + TSTEPS * TSTEPS;       // TSTEPS
  const int b = blockIdx.x, tid = threadIdx.x;
  const float* inb = input + (size_t)b * TSTEPS * NFEAT;
  for (int i = tid; i < TSTEPS * NFEAT; i += 256) in_s[i] = inb[i];
  for (int i = tid; i < TSTEPS * TSTEPS; i += 256) w2_s[i] = W2[i];
  if (tid < TSTEPS) b2_s[tid] = b2[tid];
  __syncthreads();
  float* z2b = z2 + (size_t)b * NFEAT * TPAD;
  for (int idx = tid; idx < NFEAT * TSTEPS; idx += 256) {
    const int n = idx / TSTEPS, s = idx - n * TSTEPS;
    const float* wrow = w2_s + s * TSTEPS;
    const float* icol = in_s + n;
    float acc = b2_s[s];
#pragma unroll 4
    for (int tt = 0; tt < TSTEPS; ++tt) acc = fmaf(icol[tt * NFEAT], wrow[tt], acc);
    z2b[n * TPAD + s] = acc;
  }
  // zero the pad columns (s = 99..111)
  for (int idx = tid; idx < NFEAT * (TPAD - TSTEPS); idx += 256) {
    const int n = idx / (TPAD - TSTEPS);
    const int s = TSTEPS + (idx - n * (TPAD - TSTEPS));
    z2b[n * TPAD + s] = 0.0f;
  }
}

// ---------------------------------------------------------------------------
// Kernel 3: persistent recurrent scan.  grid = 32, block = 512 (16 waves).
// ---------------------------------------------------------------------------
__global__ __launch_bounds__(NTHR) void encoder_recurrent(
    const float* __restrict__ input, const float* __restrict__ b1,
    const float* __restrict__ W3, const float* __restrict__ b3,
    const float* __restrict__ b_ih, const float* __restrict__ b_hh,
    const float* __restrict__ z2, const _Float16* __restrict__ W1h,
    const _Float16* __restrict__ Wc16, float* __restrict__ out_w,
    float* __restrict__ out_h) {
  extern __shared__ char smem[];
  float*    gs    = (float*)smem;                       // 16*1028 f32
  _Float16* hc16  = (_Float16*)(gs + BT * GSLD);        // 16*520  f16 [h|c]
  _Float16* wh16  = hc16 + BT * HCLD;                   // 16*392  f16 [w|h]
  float*    z1s   = (float*)(wh16 + BT * WHLD);         // 16*116  f32
  float*    es    = z1s + BT * TPADL;                   // 16*128  f32
  float*    biass = es + BT * NFEAT;                    // 1024
  float*    b1s   = biass + GOUT;                       // 116
  float*    w3s   = b1s + TPADL;                        // 116
  float*    rmax  = w3s + TPADL;                        // 16
  float*    rsum  = rmax + BT;                          // 16

  const int tid  = threadIdx.x;
  const int wave = tid >> 5;
  const int lane = tid & 31;
  const int b0   = blockIdx.x << 4;

  // ----- init: zero state, fold biases, warm weight caches -----
  for (int i = tid; i < BT * HCLD; i += NTHR) hc16[i] = (_Float16)0.0f;
  for (int i = tid; i < BT * WHLD; i += NTHR) wh16[i] = (_Float16)0.0f;
  for (int i = tid; i < GOUT; i += NTHR) biass[i] = b_ih[i] + b_hh[i];
  if (tid < TPADL) {
    b1s[tid] = (tid < TSTEPS) ? b1[tid] : 0.0f;
    w3s[tid] = (tid < TSTEPS) ? W3[tid] : 0.0f;   // pad of w3 = 0 kills pad cols
  }
  const float b3v = b3[0];
  // global_prefetch_b8: pull the (L2-resident) f16 weights toward the WGP
  for (int i = tid * 32; i < GOUT * KCAT; i += NTHR * 32)
    __builtin_prefetch(Wc16 + i, 0, 1);
  for (int i = tid * 32; i < TPAD * H2; i += NTHR * 32)
    __builtin_prefetch(W1h + i, 0, 1);

  float creg[8] = {0.f, 0.f, 0.f, 0.f, 0.f, 0.f, 0.f, 0.f};
  const int lm  = tid & 15;                  // lstm/elementwise row
  const int ljb = (tid >> 4) * 8;            // lstm j base (0..248)
  const int em  = tid >> 5;                  // attention row (0..15)
  const int en  = tid & 31;                  // attention col base
  __syncthreads();

  for (int t = 0; t < TSTEPS; ++t) {
    // ---- 1) z1 = [h|c](16x512) @ W1h^T -> 16x112, waves 0..6 ----
    if (wave < 7) {
      fragf32 acc = {0.f, 0.f, 0.f, 0.f, 0.f, 0.f, 0.f, 0.f};
      const int n0 = wave << 4;
      for (int kk = 0; kk < H2 / 32; ++kk) {
        frag16 a  = load_a_rowmajor(hc16, HCLD, kk * 32);
        frag16 bb = load_b_outmajor(W1h, H2, n0, kk * 32);
        acc = wmma_f16(a, bb, acc);
      }
      const int n = lane & 15, hs = lane >> 4;
#pragma unroll
      for (int r = 0; r < 8; ++r)
        z1s[(r + 8 * hs) * TPADL + n0 + n] = acc[r] + b1s[n0 + n];
    }
    __syncthreads();

    // ---- 2) e[m,n] = sum_t tanh(z1[m,t] + z2[m,n,t]) * w3[t] + b3 ----
    // float4 over the padded 112-wide rows (pads contribute 0 via w3).
    {
      const float4* z1v = (const float4*)(z1s + em * TPADL);
      const float4* w3v = (const float4*)w3s;
#pragma unroll
      for (int k = 0; k < 4; ++k) {
        const int n = en + 32 * k;
        const float4* z2v =
            (const float4*)(z2 + ((size_t)(b0 + em) * NFEAT + n) * TPAD);
        float acc = 0.0f;
#pragma unroll 4
        for (int q = 0; q < TPAD / 4; ++q) {
          const float4 a = z1v[q];
          const float4 b = z2v[q];
          const float4 w = w3v[q];
          acc = fmaf(w.x, tanhf(a.x + b.x), acc);
          acc = fmaf(w.y, tanhf(a.y + b.y), acc);
          acc = fmaf(w.z, tanhf(a.z + b.z), acc);
          acc = fmaf(w.w, tanhf(a.w + b.w), acc);
        }
        es[em * NFEAT + n] = acc + b3v;
      }
    }
    __syncthreads();

    // ---- 3) softmax stats over N (one wave per batch row) ----
    {
      const int m = wave;
      float mx = -3.4e38f;
      for (int n = lane; n < NFEAT; n += 32) mx = fmaxf(mx, es[m * NFEAT + n]);
      for (int off = 16; off; off >>= 1) mx = fmaxf(mx, __shfl_xor(mx, off, 32));
      float s = 0.0f;
      for (int n = lane; n < NFEAT; n += 32) s += __expf(es[m * NFEAT + n] - mx);
      for (int off = 16; off; off >>= 1) s += __shfl_xor(s, off, 32);
      if (lane == 0) { rmax[m] = mx; rsum[m] = s; }
    }
    __syncthreads();

    // ---- 4) w = softmax(e) * x_t : write output + f16 copy into [w|h] ----
    {
      const float mx = rmax[em], inv = 1.0f / rsum[em];
      const float* xrow = input + ((size_t)(b0 + em) * TSTEPS + t) * NFEAT;
      float* wout = out_w + ((size_t)(b0 + em) * TSTEPS + t) * NFEAT;
#pragma unroll
      for (int k = 0; k < 4; ++k) {
        const int n = en + 32 * k;
        const float wv = __expf(es[em * NFEAT + n] - mx) * inv * xrow[n];
        wout[n] = wv;
        wh16[em * WHLD + n] = (_Float16)wv;
      }
    }
    __syncthreads();

    // ---- 5) gates = [w|h](16x384) @ Wc16^T -> 16x1024 (+bias) ----
    for (int tile = wave; tile < GOUT / 16; tile += 16) {
      fragf32 acc = {0.f, 0.f, 0.f, 0.f, 0.f, 0.f, 0.f, 0.f};
      const int n0 = tile << 4;
      for (int kk = 0; kk < KCAT / 32; ++kk) {
        frag16 a  = load_a_rowmajor(wh16, WHLD, kk * 32);
        frag16 bb = load_b_outmajor(Wc16, KCAT, n0, kk * 32);
        acc = wmma_f16(a, bb, acc);
      }
      const int n = lane & 15, hs = lane >> 4;
#pragma unroll
      for (int r = 0; r < 8; ++r)
        gs[(r + 8 * hs) * GSLD + n0 + n] = acc[r] + biass[n0 + n];
    }
    __syncthreads();

    // ---- 6) LSTM elementwise; c in registers, h -> out + f16 state ----
    {
      float* hrow = out_h + ((size_t)(b0 + lm) * TSTEPS + t) * HID;
      const float* g = gs + lm * GSLD;
#pragma unroll
      for (int q = 0; q < 8; ++q) {
        const int j = ljb + q;
        const float ig = sigmoidf(g[j]);
        const float fg = sigmoidf(g[HID + j]);
        const float gg = tanhf(g[2 * HID + j]);
        const float og = sigmoidf(g[3 * HID + j]);
        const float c = fg * creg[q] + ig * gg;
        creg[q] = c;
        const float h = og * tanhf(c);
        hrow[j] = h;
        hc16[lm * HCLD + j]       = (_Float16)h;
        hc16[lm * HCLD + HID + j] = (_Float16)c;
        wh16[lm * WHLD + NFEAT + j] = (_Float16)h;
      }
    }
    __syncthreads();
  }
}

// ---------------------------------------------------------------------------
// Launch
// ---------------------------------------------------------------------------
extern "C" void kernel_launch(void* const* d_in, const int* in_sizes, int n_in,
                              void* d_out, int out_size, void* d_ws,
                              size_t ws_size, hipStream_t stream) {
  (void)in_sizes; (void)n_in; (void)out_size; (void)ws_size;
  const float* input = (const float*)d_in[0];
  const float* W1    = (const float*)d_in[1];
  const float* b1    = (const float*)d_in[2];
  const float* W2    = (const float*)d_in[3];
  const float* b2    = (const float*)d_in[4];
  const float* W3    = (const float*)d_in[5];
  const float* b3    = (const float*)d_in[6];
  const float* W_ih  = (const float*)d_in[7];
  const float* W_hh  = (const float*)d_in[8];
  const float* b_ih  = (const float*)d_in[9];
  const float* b_hh  = (const float*)d_in[10];

  float* out_w = (float*)d_out;                                   // (B,99,128)
  float* out_h = out_w + (size_t)BATCH * TSTEPS * NFEAT;          // (B,99,256)

  // workspace layout (256B-aligned):
  //   z2   f32 (B * NFEAT * TPAD)  = 29,360,128 B
  //   W1h  f16 (TPAD * H2)         =    114,688 B
  //   Wc16 f16 (GOUT * KCAT)       =    786,432 B
  char* ws = (char*)d_ws;
  float*    z2   = (float*)ws;
  _Float16* W1h  = (_Float16*)(ws + 29360128);
  _Float16* Wc16 = (_Float16*)(ws + 29360128 + 114688);

  {
    const int total = TPAD * H2 + GOUT * KCAT;  // 450,560
    prep_weights<<<(total + 255) / 256, 256, 0, stream>>>(W1, W_ih, W_hh, W1h,
                                                          Wc16);
  }
  {
    const size_t shmem = (size_t)(TSTEPS * NFEAT + TSTEPS * TSTEPS + TSTEPS) *
                         sizeof(float);  // 90,288 B
    z2_kernel<<<BATCH, 256, shmem, stream>>>(input, W2, b2, z2);
  }
  {
    const size_t shmem =
        (size_t)BT * GSLD * 4 + (size_t)BT * HCLD * 2 + (size_t)BT * WHLD * 2 +
        (size_t)BT * TPADL * 4 + (size_t)BT * NFEAT * 4 + GOUT * 4 +
        TPADL * 4 + TPADL * 4 + BT * 4 + BT * 4;  // 115,744 B
    encoder_recurrent<<<BATCH / BT, NTHR, shmem, stream>>>(
        input, b1, W3, b3, b_ih, b_hh, z2, W1h, Wc16, out_w, out_h);
  }
}
